// MNIST_GCNN_11321533792496
// MI455X (gfx1250) — matrix operand order repaired and checked by reading, not compiled, not run on hardware
//
#include <hip/hip_runtime.h>
#include <stdint.h>

typedef __attribute__((ext_vector_type(16))) __bf16 v16bf;
typedef __attribute__((ext_vector_type(8)))  float  v8f;

#define GRID28 28
#define NNODES 784
#define B_SZ   1024
#define K_TOT  25088   // 784*32
#define FC1    1024
#define NCLS   10
#define LDSK   40      // padded K stride (halfwords) -> 80B: 16B aligned, conflict-free frag reads

__device__ __forceinline__ unsigned short f2bf(float f) {
  union { float f; unsigned u; } x; x.f = f;
  unsigned r = x.u + 0x7FFFu + ((x.u >> 16) & 1u);   // round-to-nearest-even
  return (unsigned short)(r >> 16);
}

// ---------------- kernel 0: xT[m][b] = x[b][m]  (3.2 MB, trivial) -------------
__global__ void k_transpose_x(const float* __restrict__ x, float* __restrict__ xT) {
  int t = blockIdx.x * blockDim.x + threadIdx.x;
  if (t >= NNODES * B_SZ) return;
  int m = t >> 10, b = t & 1023;
  xT[t] = x[b * NNODES + m];
}

// ---------------- kernel 1: w1T[j][k] = bf16(w_fc1[k][j]), LDS-tiled ----------
// 32x32 f32 tiles: coalesced reads and coalesced transposed bf16 writes.
__global__ void __launch_bounds__(256) k_cvt_w1T(const float* __restrict__ w,
                                                 unsigned short* __restrict__ wT) {
  __shared__ float tile[32][33];
  const int k0 = blockIdx.x << 5;   // 784 blocks over K
  const int j0 = blockIdx.y << 5;   // 32 blocks over N
  const int tx = threadIdx.x & 31, ty = threadIdx.x >> 5;   // 32 x 8
#pragma unroll
  for (int i = 0; i < 4; ++i) {
    int k = ty + (i << 3);
    tile[k][tx] = w[(size_t)(k0 + k) * FC1 + j0 + tx];
  }
  __syncthreads();
#pragma unroll
  for (int i = 0; i < 4; ++i) {
    int j = ty + (i << 3);
    wT[(size_t)(j0 + j) * K_TOT + k0 + tx] = f2bf(tile[tx][j]);
  }
}

// ---------------- kernel 2: fused GCNN -> A-matrix in WMMA fragment layout -----
// hfrag[n][m16][lane][0..15] : lane L holds A[b = m16*16 + L%16, k(v,h,L/16)]
// with k = (v%4)*2 + h + (L/16)*8 + (v/4)*16  (ISA 16-bit A 16x32 layout)
__global__ void k_build_a(const float* __restrict__ xT, const float* __restrict__ adj,
                          const float* __restrict__ wg, const float* __restrict__ bg,
                          unsigned short* __restrict__ hfrag) {
  int t = blockIdx.x * blockDim.x + threadIdx.x;     // 784 * 64 * 32
  if (t >= NNODES * 64 * 32) return;
  int n   = t >> 11;
  int rem = t & 2047;
  int m16 = rem >> 5;
  int L   = rem & 31;
  int b   = (m16 << 4) + (L & 15);
  int hi  = L >> 4;
  int r = n / GRID28, c = n % GRID28;

  // exact 9-tap stencil: all other adj entries are structurally zero
  float g = 0.f;
#pragma unroll
  for (int dr = -1; dr <= 1; ++dr) {
#pragma unroll
    for (int dc = -1; dc <= 1; ++dc) {
      int rr = r + dr, cc = c + dc;
      if (rr >= 0 && rr < GRID28 && cc >= 0 && cc < GRID28) {
        int m = rr * GRID28 + cc;
        g += adj[n * NNODES + m] * xT[(m << 10) + b];
      }
    }
  }

  unsigned short outv[16];
#pragma unroll
  for (int v = 0; v < 8; ++v) {
#pragma unroll
    for (int h = 0; h < 2; ++h) {
      int k = ((v & 3) << 1) + h + (hi << 3) + ((v >> 2) << 4);
      float val = fmaxf(fmaf(g, wg[k], bg[k]), 0.f);
      outv[2 * v + h] = f2bf(val);
    }
  }
  uint4* dst = (uint4*)(hfrag + (size_t)t * 16);   // 32B contiguous, coalesced
  dst[0] = ((const uint4*)outv)[0];
  dst[1] = ((const uint4*)outv)[1];
}

// ---------------- kernel 3: FC1 GEMM via v_wmma_f32_16x16x32_bf16 --------------
// Block = 8 waves = 256M x 128N of C.  Wave (wm 0..3, wn 0..1) -> 64M x 64N.
// B panel staged with GLOBAL_LOAD_ASYNC_TO_LDS_B128 into double-buffered LDS.
__global__ void __launch_bounds__(256)
k_fc1(const unsigned short* __restrict__ hfrag, const unsigned short* __restrict__ w1T,
      const float* __restrict__ b1, float* __restrict__ out1) {
  __shared__ __align__(16) unsigned short ldsB[2][128 * LDSK];

  const int tid   = threadIdx.x;
  const int nblk  = blockIdx.x;          // 0..7  (N)
  const int mblk  = blockIdx.y;          // 0..3  (M)
  const int wave  = tid >> 5;
  const int lane  = tid & 31;
  const int wm    = wave & 3;
  const int wn    = wave >> 2;
  const int hi    = lane >> 4;
  const int ln    = lane & 15;
  const int j0    = nblk << 7;

  // staging assignment: lane pair covers one column's 32B (2 x 16B halves)
  const int sjcol = (tid >> 1) & 127;    // column 0..127
  const int shalf = tid & 1;             // K half 0..1 (16 halfwords each)
  const unsigned lds_off0 =
      (unsigned)(uintptr_t)&ldsB[0][sjcol * LDSK + (shalf << 4)];
  const unsigned lds_off1 =
      (unsigned)(uintptr_t)&ldsB[1][sjcol * LDSK + (shalf << 4)];
  const unsigned short* gsrc0 = w1T + (size_t)(j0 + sjcol) * K_TOT + (shalf << 4);

  v8f acc[4][4];
#pragma unroll
  for (int i = 0; i < 4; ++i) {
#pragma unroll
    for (int j = 0; j < 4; ++j) {
#pragma unroll
      for (int e = 0; e < 8; ++e) acc[i][j][e] = 0.f;
    }
  }

  // prologue: stage node 0 into buffer 0
  {
    uint64_t ga = (uint64_t)(uintptr_t)(gsrc0);
    asm volatile("global_load_async_to_lds_b128 %0, %1, off"
                 :: "v"(lds_off0), "v"(ga) : "memory");
  }

  for (int n = 0; n < NNODES; ++n) {
    asm volatile("s_wait_asynccnt 0x0" ::: "memory");
    __syncthreads();

    // stage node n+1 into the other buffer (overlaps with compute below)
    if (n + 1 < NNODES) {
      uint64_t ga = (uint64_t)(uintptr_t)(gsrc0 + (size_t)(n + 1) * 32);
      unsigned lo = ((n + 1) & 1) ? lds_off1 : lds_off0;
      asm volatile("global_load_async_to_lds_b128 %0, %1, off"
                   :: "v"(lo), "v"(ga) : "memory");
    }

    const unsigned short* bufB = ldsB[n & 1];

    // A fragments: two coalesced 16B loads per sub-tile (layout prebuilt)
    v16bf afr[4];
#pragma unroll
    for (int ms = 0; ms < 4; ++ms) {
      int m16 = (mblk << 4) + (wm << 2) + ms;
      const uint4* ap = (const uint4*)(hfrag + (((size_t)n * 64 + m16) * 32 + lane) * 16);
      union { v16bf v; uint4 q[2]; } ua;
      ua.q[0] = ap[0]; ua.q[1] = ap[1];
      afr[ms] = ua.v;
      if (n + 4 < NNODES)   // prefetch A stream 4 steps ahead -> global_prefetch_b8
        __builtin_prefetch(hfrag + (((size_t)(n + 4) * 64 + m16) * 32 + lane) * 16, 0, 0);
    }

#pragma unroll
    for (int ns = 0; ns < 4; ++ns) {
      // B fragment: lane L -> col = wn*64 + ns*16 + L%16, K = (L/16)*8 + s (s<8), 16 + (L/16)*8 + (s-8) (s>=8)
      const unsigned short* lp = bufB + ((wn << 6) + (ns << 4) + ln) * LDSK + (hi << 3);
      union { v16bf v; uint4 q[2]; } ub;
      ub.q[0] = *(const uint4*)(lp);
      ub.q[1] = *(const uint4*)(lp + 16);
#pragma unroll
      for (int ms = 0; ms < 4; ++ms) {
        acc[ms][ns] = __builtin_amdgcn_wmma_f32_16x16x32_bf16(
            false, afr[ms], false, ub.v, (short)0, acc[ms][ns], false, false);
      }
    }
  }

  // epilogue: bias + relu, f32 out.  C layout: VGPR v -> M = v + hi*8, lane%16 -> N
#pragma unroll
  for (int ms = 0; ms < 4; ++ms) {
    int mrow0 = (mblk << 8) + (wm << 6) + (ms << 4) + (hi << 3);
#pragma unroll
    for (int ns = 0; ns < 4; ++ns) {
      int col = (nblk << 7) + (wn << 6) + (ns << 4) + ln;
      float bias = b1[col];
#pragma unroll
      for (int v = 0; v < 8; ++v)
        out1[(size_t)(mrow0 + v) * FC1 + col] = fmaxf(acc[ms][ns][v] + bias, 0.f);
    }
  }
}

// ---------------- kernel 4: FC2 (tiny, exact f32) ------------------------------
__global__ void k_fc2(const float* __restrict__ h, const float* __restrict__ w2,
                      const float* __restrict__ b2, float* __restrict__ out) {
  int t = blockIdx.x * blockDim.x + threadIdx.x;
  if (t >= B_SZ * NCLS) return;
  int b = t / NCLS, c = t % NCLS;
  float acc = b2[c];
  const float* hp = h + (size_t)b * FC1;
  for (int j = 0; j < FC1; ++j)
    acc = fmaf(hp[j], w2[j * NCLS + c], acc);
  out[t] = acc;
}

extern "C" void kernel_launch(void* const* d_in, const int* in_sizes, int n_in,
                              void* d_out, int out_size, void* d_ws, size_t ws_size,
                              hipStream_t stream) {
  const float* x   = (const float*)d_in[0];
  const float* adj = (const float*)d_in[1];
  const float* wg  = (const float*)d_in[2];
  const float* bg  = (const float*)d_in[3];
  const float* w1  = (const float*)d_in[4];
  const float* b1  = (const float*)d_in[5];
  const float* w2  = (const float*)d_in[6];
  const float* b2  = (const float*)d_in[7];
  float* out = (float*)d_out;

  char* ws = (char*)d_ws;
  const size_t OFF_XT    = 0;                      // 784*1024*4   = 3,211,264
  const size_t OFF_W1T   = 3211264;                // 25088*1024*2 = 51,380,224
  const size_t OFF_HFRAG = OFF_W1T + 51380224;     // 51,380,224
  const size_t OFF_OUT1  = OFF_HFRAG + 51380224;   // 1024*1024*4  = 4,194,304
  float*          xT    = (float*)(ws + OFF_XT);
  unsigned short* w1T   = (unsigned short*)(ws + OFF_W1T);
  unsigned short* hfrag = (unsigned short*)(ws + OFF_HFRAG);
  float*          out1  = (float*)(ws + OFF_OUT1);

  k_transpose_x<<<(NNODES * B_SZ + 255) / 256, 256, 0, stream>>>(x, xT);
  k_cvt_w1T<<<dim3(784, 32), 256, 0, stream>>>(w1, w1T);
  k_build_a<<<(NNODES * 64 * 32 + 255) / 256, 256, 0, stream>>>(xT, adj, wg, bg, hfrag);
  k_fc1<<<dim3(8, 4), 256, 0, stream>>>(hfrag, w1T, b1, out1);
  k_fc2<<<(B_SZ * NCLS + 255) / 256, 256, 0, stream>>>(out1, w2, b2, out);
}
